// InterFrameAttention_1_90855738180249
// MI455X (gfx1250) — matrix operand order repaired and checked by reading, not compiled, not run on hardware
//
#include <hip/hip_runtime.h>

// ---------------------------------------------------------------------------
// InterFrameAttention for MI455X (gfx1250): bf16 WMMA pipeline
//   - 128x128x32 LDS-tiled GEMMs, 8 wave32 waves, double-buffered staging
//   - GLOBAL_LOAD_ASYNC_TO_LDS_B128 staging path (ASYNCcnt) when available
//   - k^T v / k^T cor reductions via transposed LDS tiles + f32 atomics
// ---------------------------------------------------------------------------
typedef unsigned short u16;
typedef __bf16 bf16_t;
typedef bf16_t v16bf __attribute__((ext_vector_type(16)));
typedef float  v8f   __attribute__((ext_vector_type(8)));
typedef int    v4i   __attribute__((vector_size(16)));   // matches builtin param

union Frag { v16bf v; uint4 q[2]; };
union Pack8 { uint4 q; u16 h[8]; };

#define NB   8
#define NN   8192
#define NC   512
#define NHD  8
#define HD   64
#define MD   256
#define MDH  32
#define MROWS (NB * NN)          // 65536
#define EPSV 1e-5f

#if defined(__has_builtin)
# if __has_builtin(__builtin_amdgcn_global_load_async_to_lds_b128)
#  define HAVE_ASYNC_B128 1
# endif
# if __has_builtin(__builtin_amdgcn_s_wait_asynccnt)
#  define HAVE_WAIT_ASYNC 1
# endif
#endif

__device__ __forceinline__ u16 f2bf(float f) {
  unsigned u = __float_as_uint(f);
  unsigned r = u + 0x7FFFu + ((u >> 16) & 1u);
  return (u16)(r >> 16);
}
__device__ __forceinline__ float bf2f(u16 h) {
  return __uint_as_float(((unsigned)h) << 16);
}
__device__ __forceinline__ float gelu_exact(float x) {
  return 0.5f * x * (1.0f + erff(x * 0.70710678118654752f));
}
__device__ __forceinline__ v8f wmma_bf16(const Frag& a, const Frag& b, v8f c) {
  return __builtin_amdgcn_wmma_f32_16x16x32_bf16(false, a.v, false, b.v,
                                                 (short)0, c, false, false);
}

#if defined(HAVE_ASYNC_B128)
// 16B async copy global->LDS per lane (tracked by ASYNCcnt).
// Builtin signature (from compiler diagnostic): param0 = v4i __device__* (AS1).
// Flat->named addrspace via uintptr_t round-trip: LDS addressing discards
// upper bits (ISA 10.1.1); global aperture is identity-mapped.
typedef __attribute__((address_space(1))) v4i* gas_b128;
typedef __attribute__((address_space(3))) v4i* las_b128;
__device__ __forceinline__ void async_b128(u16* lds_dst, const u16* gsrc) {
  __builtin_amdgcn_global_load_async_to_lds_b128(
      (gas_b128)(uintptr_t)gsrc, (las_b128)(uintptr_t)lds_dst, 0, 0);
}
__device__ __forceinline__ void wait_async0() {
# if defined(HAVE_WAIT_ASYNC)
  __builtin_amdgcn_s_wait_asynccnt(0);
# else
  asm volatile("s_wait_asynccnt 0x0" ::: "memory");
# endif
}
#endif

// ---------------------------------------------------------------------------
// utility kernels
// ---------------------------------------------------------------------------
__global__ __launch_bounds__(256) void zero_f32_kernel(float* p, long n) {
  long i = (long)blockIdx.x * 256 + threadIdx.x;
  if (i < n) p[i] = 0.0f;
}

__global__ __launch_bounds__(256) void cvt_bf16_kernel(const float* __restrict__ in,
                                                       u16* __restrict__ out, long n4) {
  long i = (long)blockIdx.x * 256 + threadIdx.x;
  if (i >= n4) return;
  float4 v = ((const float4*)in)[i];
  Pack8 pk;
  pk.h[0] = f2bf(v.x); pk.h[1] = f2bf(v.y); pk.h[2] = f2bf(v.z); pk.h[3] = f2bf(v.w);
  uint2 w; w.x = pk.q.x; w.y = pk.q.y;
  ((uint2*)out)[i] = w;
}

// cor_embed_ = cor @ cor_w.T + cor_b      (f32, B*N x MD)
__global__ __launch_bounds__(256) void cor_embed_kernel(const float* __restrict__ cor,
                                                        const float* __restrict__ cw,
                                                        const float* __restrict__ cb,
                                                        float* __restrict__ corE) {
  size_t idx = (size_t)blockIdx.x * 256 + threadIdx.x;   // < 65536*256
  size_t row = idx >> 8;
  int    md  = (int)(idx & 255);
  float c0 = cor[row * 2 + 0], c1 = cor[row * 2 + 1];
  corE[idx] = c0 * cw[md * 2 + 0] + c1 * cw[md * 2 + 1] + cb[md];
}

// ---------------------------------------------------------------------------
// Per-head LayerNorm (torch clone: unbiased std, /(std+eps)); in-place on
// bf16 kv buffer (row stride 2C = 1024; 16 groups of 64 per row: 8 k, 8 v).
// One wave32 per group, 2 elements/lane, shuffle reductions.
// ---------------------------------------------------------------------------
__global__ __launch_bounds__(256) void ln_kv_kernel(u16* __restrict__ kv,
                                                    const float* __restrict__ klnw,
                                                    const float* __restrict__ klnb,
                                                    const float* __restrict__ vlnw,
                                                    const float* __restrict__ vlnb) {
  int lane = threadIdx.x & 31, wave = threadIdx.x >> 5;
  size_t gw  = (size_t)blockIdx.x * 8 + wave;     // 65536*16 groups
  size_t row = gw >> 4;
  int    g   = (int)(gw & 15);
  u16* p = kv + row * 1024 + g * 64;
  float x0 = bf2f(p[lane]), x1 = bf2f(p[lane + 32]);
  float s = x0 + x1, s2 = x0 * x0 + x1 * x1;
#pragma unroll
  for (int o = 16; o; o >>= 1) { s += __shfl_xor(s, o, 32); s2 += __shfl_xor(s2, o, 32); }
  float mean = s * (1.0f / 64.0f);
  float var  = (s2 - 64.0f * mean * mean) * (1.0f / 63.0f);
  float inv  = 1.0f / (sqrtf(fmaxf(var, 0.0f)) + EPSV);
  const float *wv, *bv; int h;
  if (g < 8) { h = g;     wv = klnw; bv = klnb; }
  else       { h = g - 8; wv = vlnw; bv = vlnb; }
  p[lane]      = f2bf(wv[h * 64 + lane]      * (x0 - mean) * inv + bv[h * 64 + lane]);
  p[lane + 32] = f2bf(wv[h * 64 + lane + 32] * (x1 - mean) * inv + bv[h * 64 + lane + 32]);
}

// cor LN: groups of MDH=32 per head, one wave per group; f32 in -> bf16 out
__global__ __launch_bounds__(256) void ln_cor_kernel(const float* __restrict__ corE,
                                                     u16* __restrict__ corb,
                                                     const float* __restrict__ w,
                                                     const float* __restrict__ b) {
  int lane = threadIdx.x & 31, wave = threadIdx.x >> 5;
  size_t gw  = (size_t)blockIdx.x * 8 + wave;     // 65536*8 groups
  size_t row = gw >> 3;
  int    h   = (int)(gw & 7);
  float x = corE[row * 256 + h * 32 + lane];
  float s = x, s2 = x * x;
#pragma unroll
  for (int o = 16; o; o >>= 1) { s += __shfl_xor(s, o, 32); s2 += __shfl_xor(s2, o, 32); }
  float mean = s * (1.0f / 32.0f);
  float var  = (s2 - 32.0f * mean * mean) * (1.0f / 31.0f);
  float inv  = 1.0f / (sqrtf(fmaxf(var, 0.0f)) + EPSV);
  corb[row * 256 + h * 32 + lane] =
      f2bf(w[h * 32 + lane] * (x - mean) * inv + b[h * 32 + lane]);
}

// ---------------------------------------------------------------------------
// Main bf16 WMMA GEMM: out = epi(A[M,K] @ W[Nn,K]^T + bias)  (+resid)
// Block tile 128x128x32, 8 waves (2x4), wave tile 64x32 (4x2 WMMA tiles).
// Double-buffered LDS staging; async global->LDS when the toolchain has it.
// ---------------------------------------------------------------------------
#define BM 128
#define BN 128
#define BK 32
#define LDT 40   // padded LDS row stride in halves (16B-aligned rows)

template <bool GELU, bool RESID>
__global__ __launch_bounds__(256) void gemm_bf16_kernel(
    const u16* __restrict__ A, const u16* __restrict__ W,
    const float* __restrict__ bias, const float* __restrict__ resid,
    u16* __restrict__ outbf, float* __restrict__ outf,
    int M, int Nn, int K) {
  __shared__ u16 sA[2][BM * LDT];
  __shared__ u16 sB[2][BN * LDT];
  const int tid = threadIdx.x, lane = tid & 31, wave = tid >> 5;
  const int wm = wave >> 2, wn = wave & 3;
  const int m0 = blockIdx.y * BM, n0 = blockIdx.x * BN;

  // each thread stages 2 chunks of 16B per matrix per K-step
  const int srow0 = tid >> 2,           skc0 = (tid & 3) * 8;
  const int srow1 = (tid + 256) >> 2,   skc1 = ((tid + 256) & 3) * 8;

  v8f acc[4][2] = {};
  const int nk = K / BK;

#if defined(HAVE_ASYNC_B128)
  // prologue: async-stage tile 0 into buffer 0
  {
    async_b128(&sA[0][srow0 * LDT + skc0], A + (size_t)(m0 + srow0) * K + skc0);
    async_b128(&sB[0][srow0 * LDT + skc0], W + (size_t)(n0 + srow0) * K + skc0);
    async_b128(&sA[0][srow1 * LDT + skc1], A + (size_t)(m0 + srow1) * K + skc1);
    async_b128(&sB[0][srow1 * LDT + skc1], W + (size_t)(n0 + srow1) * K + skc1);
  }
  wait_async0();
  __syncthreads();
#else
  // prologue: sync-stage tile 0
  {
    uint4 a0 = *(const uint4*)(A + (size_t)(m0 + srow0) * K + skc0);
    uint4 b0 = *(const uint4*)(W + (size_t)(n0 + srow0) * K + skc0);
    uint4 a1 = *(const uint4*)(A + (size_t)(m0 + srow1) * K + skc1);
    uint4 b1 = *(const uint4*)(W + (size_t)(n0 + srow1) * K + skc1);
    *(uint4*)(&sA[0][srow0 * LDT + skc0]) = a0;
    *(uint4*)(&sB[0][srow0 * LDT + skc0]) = b0;
    *(uint4*)(&sA[0][srow1 * LDT + skc1]) = a1;
    *(uint4*)(&sB[0][srow1 * LDT + skc1]) = b1;
  }
  __syncthreads();
#endif

  for (int kt = 0; kt < nk; ++kt) {
    const int cur = kt & 1, nxt = cur ^ 1;
    const int k1 = (kt + 1) * BK;
    const bool more = (kt + 1 < nk);

#if defined(HAVE_ASYNC_B128)
    if (more) {   // kick next tile into the other buffer; DMA overlaps WMMAs
      async_b128(&sA[nxt][srow0 * LDT + skc0], A + (size_t)(m0 + srow0) * K + k1 + skc0);
      async_b128(&sB[nxt][srow0 * LDT + skc0], W + (size_t)(n0 + srow0) * K + k1 + skc0);
      async_b128(&sA[nxt][srow1 * LDT + skc1], A + (size_t)(m0 + srow1) * K + k1 + skc1);
      async_b128(&sB[nxt][srow1 * LDT + skc1], W + (size_t)(n0 + srow1) * K + k1 + skc1);
    }
#else
    uint4 ra0, rb0, ra1, rb1;
    if (more) {   // global loads in flight across the WMMA phase
      __builtin_prefetch(A + (size_t)(m0 + srow0) * K + k1 + BK, 0, 1);
      __builtin_prefetch(W + (size_t)(n0 + srow0) * K + k1 + BK, 0, 1);
      ra0 = *(const uint4*)(A + (size_t)(m0 + srow0) * K + k1 + skc0);
      rb0 = *(const uint4*)(W + (size_t)(n0 + srow0) * K + k1 + skc0);
      ra1 = *(const uint4*)(A + (size_t)(m0 + srow1) * K + k1 + skc1);
      rb1 = *(const uint4*)(W + (size_t)(n0 + srow1) * K + k1 + skc1);
    }
#endif

    // fragments from current buffer + 8 WMMAs
    Frag a[4], b[2];
    const int kb  = (lane >> 4) * 16;   // A: K {0..15} / {16..31}
    const int kb2 = (lane >> 4) * 16;   // B: K {0..15} / {16..31}
#pragma unroll
    for (int i = 0; i < 4; ++i) {
      int row = wm * 64 + i * 16 + (lane & 15);
      a[i].q[0] = *(const uint4*)(&sA[cur][row * LDT + kb]);
      a[i].q[1] = *(const uint4*)(&sA[cur][row * LDT + kb + 8]);
    }
#pragma unroll
    for (int j = 0; j < 2; ++j) {
      int col = wn * 32 + j * 16 + (lane & 15);
      b[j].q[0] = *(const uint4*)(&sB[cur][col * LDT + kb2]);
      b[j].q[1] = *(const uint4*)(&sB[cur][col * LDT + kb2 + 8]);
    }
#pragma unroll
    for (int i = 0; i < 4; ++i)
#pragma unroll
      for (int j = 0; j < 2; ++j) acc[i][j] = wmma_bf16(a[i], b[j], acc[i][j]);

    if (more) {
#if defined(HAVE_ASYNC_B128)
      wait_async0();     // our async writes to lds[nxt] landed
      __syncthreads();   // everyone's landed; lds[nxt] ready, lds[cur] reusable
#else
      *(uint4*)(&sA[nxt][srow0 * LDT + skc0]) = ra0;
      *(uint4*)(&sB[nxt][srow0 * LDT + skc0]) = rb0;
      *(uint4*)(&sA[nxt][srow1 * LDT + skc1]) = ra1;
      *(uint4*)(&sB[nxt][srow1 * LDT + skc1]) = rb1;
      __syncthreads();
#endif
    }
  }

#pragma unroll
  for (int i = 0; i < 4; ++i)
#pragma unroll
    for (int j = 0; j < 2; ++j) {
      int gcol = n0 + wn * 32 + j * 16 + (lane & 15);
      float bs = bias[gcol];
#pragma unroll
      for (int r = 0; r < 8; ++r) {
        int grow = m0 + wm * 64 + i * 16 + (lane >> 4) * 8 + r;
        float v = acc[i][j][r] + bs;
        if (GELU) v = gelu_exact(v);
        size_t off = (size_t)grow * Nn + gcol;
        if (RESID) v += resid[off];
        if (outbf) outbf[off] = f2bf(v);
        if (outf)  outf[off]  = v;
      }
    }
}

// ---------------------------------------------------------------------------
// ctx_kernel: kv_ctx[bh] = k^T v / N (64x64), kc_ctx[bh] = k^T cor / N (64x32)
// grid (N-chunks=8, B*NH=64); tiles staged TRANSPOSED in LDS (d-major) so
// WMMA A fragments contract over n. Partial sums merged via f32 atomics.
// ---------------------------------------------------------------------------
#define CLDT 40
__global__ __launch_bounds__(256) void ctx_kernel(const u16* __restrict__ kv,
                                                  const u16* __restrict__ corb,
                                                  float* __restrict__ kvctx,
                                                  float* __restrict__ kcctx) {
  __shared__ u16 sK[64 * CLDT];   // [d][n]
  __shared__ u16 sV[64 * CLDT];   // [e][n]
  __shared__ u16 sC[32 * CLDT];   // [m][n]
  const int tid = threadIdx.x, lane = tid & 31, wave = tid >> 5;
  const int bh = blockIdx.y, b = bh >> 3, h = bh & 7;
  const int n0 = blockIdx.x * 1024;
  const int dt = wave >> 1, sel = wave & 1;

  v8f acc[2] = {};
  v8f accc   = {};

  for (int ns = 0; ns < 1024; ns += 32) {
    if (ns) __syncthreads();
    {
      int nrow = tid >> 3, dg = (tid & 7) * 8;
      size_t rb = (size_t)(b * NN + n0 + ns + nrow) * 1024;
      Pack8 pk, pv;
      pk.q = *(const uint4*)(kv + rb + h * 64 + dg);
      pv.q = *(const uint4*)(kv + rb + 512 + h * 64 + dg);
#pragma unroll
      for (int i = 0; i < 8; ++i) {
        sK[(dg + i) * CLDT + nrow] = pk.h[i];
        sV[(dg + i) * CLDT + nrow] = pv.h[i];
      }
      if (tid < 128) {
        int nr2 = tid >> 2, dg2 = (tid & 3) * 8;
        Pack8 pc;
        pc.q = *(const uint4*)(corb + (size_t)(b * NN + n0 + ns + nr2) * 256 +
                               h * 32 + dg2);
#pragma unroll
        for (int i = 0; i < 8; ++i) sC[(dg2 + i) * CLDT + nr2] = pc.h[i];
      }
    }
    __syncthreads();

    const int kb = (lane >> 4) * 16, kb2 = (lane >> 4) * 16;
    Frag a;
    int drow = dt * 16 + (lane & 15);
    a.q[0] = *(const uint4*)(sK + drow * CLDT + kb);
    a.q[1] = *(const uint4*)(sK + drow * CLDT + kb + 8);
#pragma unroll
    for (int j = 0; j < 2; ++j) {
      int e = sel * 32 + j * 16 + (lane & 15);
      Frag bf;
      bf.q[0] = *(const uint4*)(sV + e * CLDT + kb2);
      bf.q[1] = *(const uint4*)(sV + e * CLDT + kb2 + 8);
      acc[j] = wmma_bf16(a, bf, acc[j]);
    }
    {
      int mc = sel * 16 + (lane & 15);
      Frag bf;
      bf.q[0] = *(const uint4*)(sC + mc * CLDT + kb2);
      bf.q[1] = *(const uint4*)(sC + mc * CLDT + kb2 + 8);
      accc = wmma_bf16(a, bf, accc);
    }
  }

  const float sc = 1.0f / (float)NN;
#pragma unroll
  for (int j = 0; j < 2; ++j)
#pragma unroll
    for (int r = 0; r < 8; ++r) {
      int d = dt * 16 + (lane >> 4) * 8 + r;
      int e = sel * 32 + j * 16 + (lane & 15);
      atomicAdd(&kvctx[(size_t)bh * 4096 + d * 64 + e], acc[j][r] * sc);
    }
#pragma unroll
  for (int r = 0; r < 8; ++r) {
    int d  = dt * 16 + (lane >> 4) * 8 + r;
    int mc = sel * 16 + (lane & 15);
    atomicAdd(&kcctx[(size_t)bh * 2048 + d * 32 + mc], accc[r] * sc);
  }
}

// ---------------------------------------------------------------------------
// qctx_kernel: x_att = q @ kv_ctx (+x1 -> xres bf16),
//              c_rev = q @ kc_ctx (-cor_embed_ -> m bf16).  K = 64.
// grid (N/128=64, B*NH=64); ctx cached once in LDS transposed ([col][k]).
// ---------------------------------------------------------------------------
__global__ __launch_bounds__(256) void qctx_kernel(const u16* __restrict__ qbf,
                                                   const float* __restrict__ kvctx,
                                                   const float* __restrict__ kcctx,
                                                   const float* __restrict__ x1,
                                                   const float* __restrict__ corE,
                                                   u16* __restrict__ xres,
                                                   u16* __restrict__ mout) {
  __shared__ u16 sKV[64 * 72];   // [e][k]
  __shared__ u16 sKC[32 * 72];   // [m][k]
  const int tid = threadIdx.x, lane = tid & 31, wave = tid >> 5;
  const int bh = blockIdx.y, b = bh >> 3, h = bh & 7;
  const int n0 = blockIdx.x * 128;

#pragma unroll
  for (int i = 0; i < 16; ++i) {
    int idx = tid * 16 + i, d = idx >> 6, e = idx & 63;
    sKV[e * 72 + d] = f2bf(kvctx[(size_t)bh * 4096 + idx]);
  }
#pragma unroll
  for (int i = 0; i < 8; ++i) {
    int idx = tid * 8 + i, d = idx >> 5, mc = idx & 31;
    sKC[mc * 72 + d] = f2bf(kcctx[(size_t)bh * 2048 + idx]);
  }
  __syncthreads();

  v8f acc[4] = {};
  v8f accm[2] = {};
  const size_t rowbase = (size_t)(b * NN + n0 + wave * 16);

#pragma unroll
  for (int kt = 0; kt < 2; ++kt) {
    Frag a;
    const int kb = (lane >> 4) * 16;
    const u16* ap = qbf + (rowbase + (lane & 15)) * 512 + h * 64 + kt * 32;
    a.q[0] = *(const uint4*)(ap + kb);
    a.q[1] = *(const uint4*)(ap + kb + 8);
    const int kb2 = kt * 32 + (lane >> 4) * 16;
#pragma unroll
    for (int et = 0; et < 4; ++et) {
      int e = et * 16 + (lane & 15);
      Frag bf;
      bf.q[0] = *(const uint4*)(sKV + e * 72 + kb2);
      bf.q[1] = *(const uint4*)(sKV + e * 72 + kb2 + 8);
      acc[et] = wmma_bf16(a, bf, acc[et]);
    }
#pragma unroll
    for (int ct = 0; ct < 2; ++ct) {
      int mc = ct * 16 + (lane & 15);
      Frag bf;
      bf.q[0] = *(const uint4*)(sKC + mc * 72 + kb2);
      bf.q[1] = *(const uint4*)(sKC + mc * 72 + kb2 + 8);
      accm[ct] = wmma_bf16(a, bf, accm[ct]);
    }
  }

#pragma unroll
  for (int et = 0; et < 4; ++et)
#pragma unroll
    for (int r = 0; r < 8; ++r) {
      size_t row = rowbase + (lane >> 4) * 8 + r;
      int col = h * 64 + et * 16 + (lane & 15);
      xres[row * 512 + col] = f2bf(acc[et][r] + x1[row * 512 + col]);
    }
#pragma unroll
  for (int ct = 0; ct < 2; ++ct)
#pragma unroll
    for (int r = 0; r < 8; ++r) {
      size_t row = rowbase + (lane >> 4) * 8 + r;
      int col = h * 32 + ct * 16 + (lane & 15);
      mout[row * 256 + col] = f2bf(accm[ct][r] - corE[row * 256 + col]);
    }
}

// ---------------------------------------------------------------------------
// host launcher
// ---------------------------------------------------------------------------
extern "C" void kernel_launch(void* const* d_in, const int* in_sizes, int n_in,
                              void* d_out, int out_size, void* d_ws, size_t ws_size,
                              hipStream_t stream) {
  const float* x1    = (const float*)d_in[0];
  const float* x2    = (const float*)d_in[1];
  const float* cor   = (const float*)d_in[2];
  const float* q_w   = (const float*)d_in[3];
  const float* q_b   = (const float*)d_in[4];
  const float* kv_w  = (const float*)d_in[5];
  const float* kv_b  = (const float*)d_in[6];
  const float* cor_w = (const float*)d_in[7];
  const float* cor_b = (const float*)d_in[8];
  const float* kln_w = (const float*)d_in[9];
  const float* kln_b = (const float*)d_in[10];
  const float* vln_w = (const float*)d_in[11];
  const float* vln_b = (const float*)d_in[12];
  const float* crln_w = (const float*)d_in[13];
  const float* crln_b = (const float*)d_in[14];
  const float* o1_w  = (const float*)d_in[15];
  const float* o1_b  = (const float*)d_in[16];
  const float* o2_w  = (const float*)d_in[17];
  const float* o2_b  = (const float*)d_in[18];
  const float* om1_w = (const float*)d_in[19];
  const float* om1_b = (const float*)d_in[20];
  const float* om2_w = (const float*)d_in[21];
  const float* om2_b = (const float*)d_in[22];

  float* outx = (float*)d_out;                       // (B,N,C)
  float* outm = outx + (size_t)MROWS * NC;           // (B,N,MD)

  char* wp = (char*)d_ws;
  auto alloc = [&](size_t bytes) -> char* {
    char* p = wp; wp += (bytes + 255) & ~(size_t)255; return p;
  };
  u16*   x1bf  = (u16*)  alloc((size_t)MROWS * NC * 2);
  u16*   x2bf  = (u16*)  alloc((size_t)MROWS * NC * 2);
  u16*   qbf   = (u16*)  alloc((size_t)MROWS * NC * 2);
  u16*   kvbf  = (u16*)  alloc((size_t)MROWS * 2 * NC * 2);
  float* corE  = (float*)alloc((size_t)MROWS * MD * 4);
  u16*   corb  = (u16*)  alloc((size_t)MROWS * MD * 2);
  float* kvctx = (float*)alloc((size_t)64 * 4096 * 4);
  float* kcctx = (float*)alloc((size_t)64 * 2048 * 4);
  u16*   xresb = (u16*)  alloc((size_t)MROWS * NC * 2);
  u16*   h1b   = (u16*)  alloc((size_t)MROWS * NC * 2);
  u16*   mb    = (u16*)  alloc((size_t)MROWS * MD * 2);
  u16*   gb    = (u16*)  alloc((size_t)MROWS * MD * 2);
  u16*   qwb   = (u16*)  alloc((size_t)NC * NC * 2);
  u16*   kvwb  = (u16*)  alloc((size_t)2 * NC * NC * 2);
  u16*   o1wb  = (u16*)  alloc((size_t)NC * NC * 2);
  u16*   o2wb  = (u16*)  alloc((size_t)NC * NC * 2);
  u16*   om1wb = (u16*)  alloc((size_t)MD * MD * 2);
  u16*   om2wb = (u16*)  alloc((size_t)MD * MD * 2);

  auto cvt = [&](const float* src, u16* dst, long n) {
    long n4 = n / 4;
    cvt_bf16_kernel<<<dim3((unsigned)((n4 + 255) / 256)), 256, 0, stream>>>(src, dst, n4);
  };

  zero_f32_kernel<<<dim3(1024), 256, 0, stream>>>(kvctx, 64L * 4096);
  zero_f32_kernel<<<dim3(512),  256, 0, stream>>>(kcctx, 64L * 2048);

  cvt(x1,    x1bf,  (long)MROWS * NC);
  cvt(x2,    x2bf,  (long)MROWS * NC);
  cvt(q_w,   qwb,   (long)NC * NC);
  cvt(kv_w,  kvwb,  (long)2 * NC * NC);
  cvt(o1_w,  o1wb,  (long)NC * NC);
  cvt(o2_w,  o2wb,  (long)NC * NC);
  cvt(om1_w, om1wb, (long)MD * MD);
  cvt(om2_w, om2wb, (long)MD * MD);

  // q = x1 @ q_w^T + q_b   ; kv = x2 @ kv_w^T + kv_b
  gemm_bf16_kernel<false, false><<<dim3(NC / BN, MROWS / BM), 256, 0, stream>>>(
      x1bf, qwb, q_b, nullptr, qbf, nullptr, MROWS, NC, NC);
  gemm_bf16_kernel<false, false><<<dim3(2 * NC / BN, MROWS / BM), 256, 0, stream>>>(
      x2bf, kvwb, kv_b, nullptr, kvbf, nullptr, MROWS, 2 * NC, NC);

  cor_embed_kernel<<<dim3((unsigned)((size_t)MROWS * MD / 256)), 256, 0, stream>>>(
      cor, cor_w, cor_b, corE);

  ln_kv_kernel<<<dim3((unsigned)((size_t)MROWS * 16 / 8)), 256, 0, stream>>>(
      kvbf, kln_w, kln_b, vln_w, vln_b);
  ln_cor_kernel<<<dim3((unsigned)((size_t)MROWS * 8 / 8)), 256, 0, stream>>>(
      corE, corb, crln_w, crln_b);

  // kv_ctx / kc_ctx reductions over N
  ctx_kernel<<<dim3(NN / 1024, NB * NHD), 256, 0, stream>>>(kvbf, corb, kvctx, kcctx);

  // q @ ctx with fused residual / cor subtraction
  qctx_kernel<<<dim3(NN / 128, NB * NHD), 256, 0, stream>>>(
      qbf, kvctx, kcctx, x1, corE, xresb, mb);

  // main branch MLP
  gemm_bf16_kernel<true, false><<<dim3(NC / BN, MROWS / BM), 256, 0, stream>>>(
      xresb, o1wb, o1_b, nullptr, h1b, nullptr, MROWS, NC, NC);
  gemm_bf16_kernel<false, true><<<dim3(NC / BN, MROWS / BM), 256, 0, stream>>>(
      h1b, o2wb, o2_b, x1, nullptr, outx, MROWS, NC, NC);

  // motion branch MLP
  gemm_bf16_kernel<true, false><<<dim3(MD / BN, MROWS / BM), 256, 0, stream>>>(
      mb, om1wb, om1_b, nullptr, gb, nullptr, MROWS, MD, MD);
  gemm_bf16_kernel<false, false><<<dim3(MD / BN, MROWS / BM), 256, 0, stream>>>(
      gb, om2wb, om2_b, nullptr, nullptr, outm, MROWS, MD, MD);
}
